// SpatioTemporalAttention_78640851189889
// MI455X (gfx1250) — matrix operand order repaired and checked by reading, not compiled, not run on hardware
//
#include <hip/hip_runtime.h>
#include <hip/hip_bf16.h>

#define SEQ 4
#define HID 64
#define EMB 32

typedef __bf16 bf16_t;
typedef __attribute__((ext_vector_type(16))) bf16_t v16bf;
typedef __attribute__((ext_vector_type(8)))  float  v8f;

union BF16x16 { v16bf v; bf16_t h[16]; };

// order-preserving float<->uint mapping so we can use integer atomicMax
__device__ __forceinline__ unsigned encOrd(float f) {
  unsigned b = __float_as_uint(f);
  return (b & 0x80000000u) ? ~b : (b | 0x80000000u);
}
__device__ __forceinline__ float decOrd(unsigned u) {
  return (u & 0x80000000u) ? __uint_as_float(u & 0x7FFFFFFFu)
                           : __uint_as_float(~u);
}

__device__ __forceinline__ float relu(float x) {
  return __builtin_fmaxf(x, 0.0f);
}

// A-operand K index for documented 16-bit A 16x32 layout
__device__ __forceinline__ int cmapA(int i, int hi) {
  int base = hi ? 8 : 0;
  return (i < 8) ? (base + i) : (16 + base + (i - 8));
}

// init node arrays + transpose y/avg_speed into float4-per-node layout
__global__ void k_prep(const float* __restrict__ y, const float* __restrict__ avs,
                       unsigned* __restrict__ lmaxu, float* __restrict__ denom,
                       float* __restrict__ num,
                       float4* __restrict__ yT, float4* __restrict__ avsT, int n) {
  int i = blockIdx.x * blockDim.x + threadIdx.x;
  if (i >= n) return;
  lmaxu[i] = 0u; denom[i] = 0.0f; num[i] = 0.0f;
  yT[i]   = make_float4(y[0 * n + i],   y[1 * n + i],   y[2 * n + i],   y[3 * n + i]);
  avsT[i] = make_float4(avs[0 * n + i], avs[1 * n + i], avs[2 * n + i], avs[3 * n + i]);
}

// Transposed-GEMM logits kernel with static/dynamic K split.
//  chunk0 (c=0..31)  = emb_prod[0..31]                      (k-invariant)
//  chunk1 (c=32..63) = [avg_speed, onehot k, ea, bias-one]  (per-k)
//  Per 16-edge tile: base[q] = W1prod^T x prod^T  (4 WMMAs, once)
//  Per k:            acc[q]  = W1dyn^T x dyn(k)^T + base[q] (4 WMMAs)
__global__ __launch_bounds__(256) void k_logits(
    const float4* __restrict__ avsT, const float* __restrict__ emb,
    const float* __restrict__ W1,  const float* __restrict__ b1,
    const float* __restrict__ W2,  const float* __restrict__ b2,
    const int* __restrict__ src,   const int* __restrict__ dst,
    const float* __restrict__ ea,
    float* __restrict__ logits,    // layout [SEQ][nEdges]
    unsigned* __restrict__ lmaxu,
    int nNodes, int nEdges)
{
  const int lane = threadIdx.x & 31;
  const int hi   = lane >> 4;
  const int col0 = lane & 15;          // my edge within the tile (both halves)
  const int wavesTotal = (gridDim.x * blockDim.x) >> 5;
  const int waveId     = (blockIdx.x * blockDim.x + threadIdx.x) >> 5;

  const bf16_t BF_ZERO = (bf16_t)0.0f;
  const bf16_t BF_ONE  = (bf16_t)1.0f;

  // ---- loop-invariant A tiles, hid rows [16q,16q+16)
  BF16x16 Aprod[4];   // chunk0: W1 rows 6..37 (emb_prod weights)
  BF16x16 Adyn[4];    // chunk1: rows for avg_speed/onehot/ea/bias
  #pragma unroll
  for (int q = 0; q < 4; ++q) {
    int h = 16 * q + col0;
    #pragma unroll
    for (int i = 0; i < 16; ++i) {
      int c = 16 * hi + i;
      Aprod[q].h[i] = (bf16_t)W1[(6 + c) * HID + h];
    }
    #pragma unroll
    for (int i = 0; i < 16; ++i) {
      int cc = cmapA(i, hi) + 32;
      float v;
      if (cc == 32)      v = W1[0 * HID + h];        // avg_speed weight
      else if (cc <= 36) v = W1[(cc - 32) * HID + h];// onehot rows 1..4
      else if (cc == 37) v = W1[5 * HID + h];        // edge_attr weight
      else if (cc == 38) v = b1[h];                  // bias via constant-1 dim
      else               v = 0.0f;
      Adyn[q].h[i] = (bf16_t)v;
    }
  }

  // per-lane W2 at my 32 hidden indices (hid = 16*nb + j + 8*hi)
  float w2r[4][8];
  #pragma unroll
  for (int nb = 0; nb < 4; ++nb)
    #pragma unroll
    for (int j = 0; j < 8; ++j) w2r[nb][j] = W2[16 * nb + j + 8 * hi];
  const float b2v = b2[0];

  const int nTiles = nEdges >> 4;
  for (int tile = waveId; tile < nTiles; tile += wavesTotal) {
    const int eBase = tile << 4;
    const int e   = eBase + col0;      // this lane's edge
    const int s_e = src[e];
    const int d_e = dst[e];
    bf16_t eabf = (bf16_t)ea[e];
    float4 a4 = avsT[s_e];             // one 16B load, all 4 time steps
    bf16_t asbf[SEQ] = { (bf16_t)a4.x, (bf16_t)a4.y, (bf16_t)a4.z, (bf16_t)a4.w };

    // B chunk0 (static): lane<16 supplies K=0..15 -> prod dims 0..15,
    //                    lane>=16 supplies K=16..31 -> prod dims 16..31
    const float4* ps = reinterpret_cast<const float4*>(emb + (size_t)s_e * EMB);
    const float4* pd = reinterpret_cast<const float4*>(emb + (size_t)d_e * EMB);
    BF16x16 Bp;
    #pragma unroll
    for (int b = 0; b < 4; ++b) {
      float4 a = ps[hi * 4 + b], c = pd[hi * 4 + b];
      Bp.h[4 * b + 0] = (bf16_t)(a.x * c.x);
      Bp.h[4 * b + 1] = (bf16_t)(a.y * c.y);
      Bp.h[4 * b + 2] = (bf16_t)(a.z * c.z);
      Bp.h[4 * b + 3] = (bf16_t)(a.w * c.w);
    }

    // static GEMM once per tile
    v8f base[4];
    #pragma unroll
    for (int q = 0; q < 4; ++q) {
      #pragma unroll
      for (int j = 0; j < 8; ++j) base[q][j] = 0.0f;
      base[q] = __builtin_amdgcn_wmma_f32_16x16x32_bf16(
          false, Aprod[q].v, false, Bp.v, (short)0, base[q], false, false);
    }

    float lmaxLoc = -__builtin_inff();
    #pragma unroll
    for (int k = 0; k < SEQ; ++k) {
      // B chunk1 (dynamic): only 7 non-zero K dims on lanes<16
      BF16x16 Bd;
      Bd.h[0] = hi ? BF_ZERO : asbf[k];
      #pragma unroll
      for (int i = 1; i <= 4; ++i)
        Bd.h[i] = (hi || (i - 1 != k)) ? BF_ZERO : BF_ONE;
      Bd.h[5] = hi ? BF_ZERO : eabf;
      Bd.h[6] = hi ? BF_ZERO : BF_ONE;
      #pragma unroll
      for (int i = 7; i < 16; ++i) Bd.h[i] = BF_ZERO;

      v8f acc[4];
      #pragma unroll
      for (int q = 0; q < 4; ++q)
        acc[q] = __builtin_amdgcn_wmma_f32_16x16x32_bf16(
            false, Adyn[q].v, false, Bd.v, (short)0, base[q], false, false);

      // ReLU + dot with W2 over my 32 hidden values
      float sum = 0.0f;
      #pragma unroll
      for (int nb = 0; nb < 4; ++nb)
        #pragma unroll
        for (int j = 0; j < 8; ++j)
          sum = __builtin_fmaf(relu(acc[nb][j]), w2r[nb][j], sum);
      sum += __shfl_xor(sum, 16, 32);
      float lg = sum + b2v;
      lmaxLoc = __builtin_fmaxf(lmaxLoc, lg);
      if (lane < 16)
        logits[(size_t)k * nEdges + e] = lg;   // coalesced [k][e] layout
    }
    if (lane < 16)
      atomicMax(lmaxu + d_e, encOrd(lmaxLoc)); // one atomic per edge
  }
}

// scalar fallback for edges beyond the last full 16-tile (unused when E%16==0)
__global__ void k_logits_rem(
    const float4* __restrict__ avsT, const float* __restrict__ emb,
    const float* __restrict__ W1,  const float* __restrict__ b1,
    const float* __restrict__ W2,  const float* __restrict__ b2,
    const int* __restrict__ src,   const int* __restrict__ dst,
    const float* __restrict__ ea,
    float* __restrict__ logits, unsigned* __restrict__ lmaxu,
    int nNodes, int eStart, int nEdges)
{
  int e = eStart + blockIdx.x * blockDim.x + threadIdx.x;
  if (e >= nEdges) return;
  int s = src[e], d = dst[e];
  float eav = ea[e];
  float prod[EMB];
  for (int i = 0; i < EMB; ++i) prod[i] = emb[(size_t)s * EMB + i] * emb[(size_t)d * EMB + i];
  float4 a4 = avsT[s];
  float asv4[SEQ] = { a4.x, a4.y, a4.z, a4.w };
  float lmaxLoc = -__builtin_inff();
  for (int k = 0; k < SEQ; ++k) {
    float asv = asv4[k];
    float sum = 0.0f;
    for (int c = 0; c < HID; ++c) {
      float h = b1[c] + asv * W1[0 * HID + c] + W1[(1 + k) * HID + c] + eav * W1[5 * HID + c];
      for (int i = 0; i < EMB; ++i) h = __builtin_fmaf(prod[i], W1[(6 + i) * HID + c], h);
      h = relu(h);
      sum = __builtin_fmaf(h, W2[c], sum);
    }
    float lg = sum + b2[0];
    logits[(size_t)k * nEdges + e] = lg;
    lmaxLoc = __builtin_fmaxf(lmaxLoc, lg);
  }
  atomicMax(lmaxu + d, encOrd(lmaxLoc));
}

__global__ void k_fixmax(unsigned* lmaxu, float* lmaxf, int n) {
  int i = blockIdx.x * blockDim.x + threadIdx.x;
  if (i < n) {
    float f = decOrd(lmaxu[i]);
    if (!isfinite(f)) f = 0.0f;        // matches jnp.where(isfinite, lmax, 0)
    lmaxf[i] = f;
  }
}

// one thread per BASE edge: sum the 4 time-steps locally -> 2 atomics/edge,
// single float4 target load via transposed y
__global__ void k_accum(const float* __restrict__ logits, const float* __restrict__ lmaxf,
                        const int* __restrict__ src, const int* __restrict__ dst,
                        const float4* __restrict__ yT,
                        float* __restrict__ denom, float* __restrict__ num,
                        int nNodes, int nEdges)
{
  int e = blockIdx.x * blockDim.x + threadIdx.x;
  if (e >= nEdges) return;
  int s = src[e], d = dst[e];
  float lm = lmaxf[d];
  float4 yv = yT[s];
  float tgt[SEQ] = { yv.x, yv.y, yv.z, yv.w };
  float dsum = 0.0f, nsum = 0.0f;
  #pragma unroll
  for (int k = 0; k < SEQ; ++k) {
    float ex = __expf(logits[(size_t)k * nEdges + e] - lm);
    dsum += ex;
    nsum = __builtin_fmaf(ex, tgt[k], nsum);
  }
  atomicAdd(denom + d, dsum);
  atomicAdd(num + d, nsum);
}

__global__ void k_final(const float* __restrict__ num, const float* __restrict__ denom,
                        float* __restrict__ out, int n) {
  int i = blockIdx.x * blockDim.x + threadIdx.x;
  if (i < n) out[i] = num[i] / (denom[i] + 1e-16f);
}

extern "C" void kernel_launch(void* const* d_in, const int* in_sizes, int n_in,
                              void* d_out, int out_size, void* d_ws, size_t ws_size,
                              hipStream_t stream) {
  const float* y   = (const float*)d_in[0];
  const float* avs = (const float*)d_in[1];
  const float* emb = (const float*)d_in[2];
  const float* W1  = (const float*)d_in[3];
  const float* b1  = (const float*)d_in[4];
  const float* W2  = (const float*)d_in[5];
  const float* b2  = (const float*)d_in[6];
  const int*   ei  = (const int*)d_in[7];
  const float* ea  = (const float*)d_in[8];

  const int nEdges = in_sizes[8];            // edge_attr flat count = E
  const int nNodes = in_sizes[2] / EMB;
  const int* src = ei;
  const int* dst = ei + nEdges;

  char* ws = (char*)d_ws;
  size_t off = 0;
  float*    logits = (float*)(ws + off); off += (size_t)nEdges * SEQ * sizeof(float);
  unsigned* lmaxu  = (unsigned*)(ws + off); off += (size_t)nNodes * sizeof(unsigned);
  float*    denom  = (float*)(ws + off); off += (size_t)nNodes * sizeof(float);
  float*    num    = (float*)(ws + off); off += (size_t)nNodes * sizeof(float);
  off = (off + 15) & ~(size_t)15;            // align float4 arrays
  float4*   yT     = (float4*)(ws + off); off += (size_t)nNodes * sizeof(float4);
  float4*   avsT   = (float4*)(ws + off); off += (size_t)nNodes * sizeof(float4);

  k_prep<<<(nNodes + 255) / 256, 256, 0, stream>>>(y, avs, lmaxu, denom, num,
                                                   yT, avsT, nNodes);

  k_logits<<<768, 256, 0, stream>>>(avsT, emb, W1, b1, W2, b2, src, dst, ea,
                                    logits, lmaxu, nNodes, nEdges);
  int rem = nEdges & 15;
  if (rem) {
    int eStart = nEdges & ~15;
    k_logits_rem<<<(rem + 63) / 64, 64, 0, stream>>>(avsT, emb, W1, b1, W2, b2,
                                                     src, dst, ea, logits, lmaxu,
                                                     nNodes, eStart, nEdges);
  }

  k_fixmax<<<(nNodes + 255) / 256, 256, 0, stream>>>(lmaxu, (float*)lmaxu, nNodes);

  k_accum<<<(nEdges + 255) / 256, 256, 0, stream>>>(logits, (const float*)lmaxu,
                                                    src, dst, yT, denom, num,
                                                    nNodes, nEdges);

  k_final<<<(nNodes + 255) / 256, 256, 0, stream>>>(num, denom, (float*)d_out, nNodes);
}
